// SelfAttention_24404004175978
// MI455X (gfx1250) — matrix operand order repaired and checked by reading, not compiled, not run on hardware
//
#include <hip/hip_runtime.h>

// ---------------------------------------------------------------------------
// Self-attention (N=8, S=2048, E=512, H=8, D=64) + output projection,
// MI455X / gfx1250 (wave32, v_wmma_f32_16x16x32_f16, TDM tensor_load_to_lds).
//
// Kernel 0 (pack):  K,V f32[N,S,H,D] -> f16[N,H,S,D]; W -> f16; mask -> f16 {0,1}.
// Kernel 1 (attn):  flash attention with TRANSPOSED tiles:
//                     S^T = K . Q^T   (C-frag column == query == lane)
//                     O^T = V^T . P^T (epilogue = contiguous b128 stores)
//                   K tile staged into LDS by the Tensor Data Mover
//                   (TENSORcnt / s_wait_tensorcnt); V^T staged manually.
//                   1/sqrt(E)*log2(e) folded into Q -> softmax in exp2 domain.
//                   Row-sums via a "ones" WMMA; mask applied multiplicatively.
// Kernel 2 (proj):  out = ctx @ W^T; 2 A-frags per wave share each W B-frag.
// ---------------------------------------------------------------------------

typedef __attribute__((ext_vector_type(16))) _Float16 v16h;
typedef __attribute__((ext_vector_type(8)))  _Float16 v8h;
typedef __attribute__((ext_vector_type(8)))  float    v8f;
typedef unsigned int u32x4 __attribute__((ext_vector_type(4)));
typedef int          i32x4 __attribute__((ext_vector_type(4)));
typedef int          i32x8 __attribute__((ext_vector_type(8)));

#define NN 8
#define SS 2048
#define EE 512
#define HH 8
#define DD 64

// workspace layout (halves)
#define KH_OFF  0u
#define VH_OFF  8388608u     // N*H*S*D
#define CTX_OFF 16777216u
#define WH_OFF  25165824u    // + N*S*E
#define MH_OFF  25427968u    // + E*E

#define WMMA_F16(A, B, C) \
  __builtin_amdgcn_wmma_f32_16x16x32_f16(false, (A), false, (B), (short)0, (C), false, false)

__device__ __forceinline__ v16h cat8(v8h lo, v8h hi) {
  return __builtin_shufflevector(lo, hi, 0,1,2,3,4,5,6,7,8,9,10,11,12,13,14,15);
}

// TDM: async-load a 64x64 f16 tile from a [S][64] f16 tensor into LDS.
// D# per CDNA5 ISA 8.3/8.4: group0 = {flags, lds_addr, global_addr, type=2},
// group1 = {data_size=2B, tensor_dim0=64, tensor_dim1=S, tile 64x64, stride 64}.
__device__ __forceinline__ void tdm_load_tile(const _Float16* gsrc,
                                              unsigned lds_byte_addr) {
  unsigned long long ga = (unsigned long long)(uintptr_t)gsrc;
  u32x4 g0;
  g0[0] = 1u;                                              // count=1 (valid D#)
  g0[1] = lds_byte_addr;                                   // lds_addr
  g0[2] = (unsigned)(ga & 0xFFFFFFFFu);                    // global_addr[31:0]
  g0[3] = (unsigned)((ga >> 32) & 0x1FFFFFFu) | (2u << 30); // addr[56:32]|type=2
  i32x8 g1;
  g1[0] = (int)(1u << 16);        // workgroup_mask=0, data_size=1 (2 bytes)
  g1[1] = (int)(64u << 16);       // tensor_dim0 = 64 (D)
  g1[2] = (int)((unsigned)SS << 16); // tensor_dim1 = S
  g1[3] = (int)(64u << 16);       // tile_dim0 = 64
  g1[4] = 64;                     // tile_dim1 = 64, tile_dim2 = 0
  g1[5] = 64;                     // tensor_dim0_stride = 64
  g1[6] = 0;                      // stride hi / tensor_dim1_stride lo (unused, 2D)
  g1[7] = 0;
  i32x4 z4 = (i32x4){0, 0, 0, 0};
#if defined(__clang_major__) && (__clang_major__ >= 23)
  i32x8 z8 = (i32x8){0, 0, 0, 0, 0, 0, 0, 0};
  __builtin_amdgcn_tensor_load_to_lds(g0, g1, z4, z4, z8, 0);
#else
  __builtin_amdgcn_tensor_load_to_lds(g0, g1, z4, z4, 0);
#endif
}

// ---------------------------------------------------------------------------
// Kernel 0: pack K,V -> f16 [N,H,S,D]; W -> f16; mask -> f16 multiplier
__global__ __launch_bounds__(256)
void pack_kernel(const float* __restrict__ Vg, const float* __restrict__ Kg,
                 const float* __restrict__ Wg, const int* __restrict__ maskg,
                 _Float16* __restrict__ Kh, _Float16* __restrict__ Vh,
                 _Float16* __restrict__ Wh, _Float16* __restrict__ Mh) {
  const int t = blockIdx.x * 256 + threadIdx.x;
  if (blockIdx.y < 2) {
    const float* src = (blockIdx.y == 0) ? Kg : Vg;
    _Float16*    dst = (blockIdx.y == 0) ? Kh : Vh;
    const int dc = (t & 7) << 3;
    const int s  = (t >> 3) & (SS - 1);
    const int h  = (t >> 14) & (HH - 1);
    const int n  = t >> 17;
    const float* p = src + ((size_t)(n * SS + s)) * EE + h * DD + dc;
    v8h o;
#pragma unroll
    for (int i = 0; i < 8; ++i) o[i] = (_Float16)p[i];
    *(v8h*)(dst + (((size_t)(n * HH + h)) * SS + s) * DD + dc) = o;
  } else {
    if (t < (EE * EE) / 8) {
      const float* p = Wg + (size_t)t * 8;
      v8h o;
#pragma unroll
      for (int i = 0; i < 8; ++i) o[i] = (_Float16)p[i];
      *(v8h*)(Wh + (size_t)t * 8) = o;
    } else if (t < (EE * EE) / 8 + NN * SS) {
      const int i = t - (EE * EE) / 8;
      Mh[i] = maskg[i] ? (_Float16)1.0f : (_Float16)0.0f;
    }
  }
}

// ---------------------------------------------------------------------------
// Kernel 1: flash attention, transposed tiles.
// grid(S/128, H, N), block 256 = 8 waves; wave w owns 16 query rows.
__global__ __launch_bounds__(256)
void attn_fwd_kernel(const _Float16* __restrict__ Kh, const _Float16* __restrict__ Vh,
                     const float* __restrict__ Qg, const _Float16* __restrict__ Mh,
                     _Float16* __restrict__ ctx) {
  __shared__ _Float16 Kl[64 * 64];        // K tile  [key][d]  (TDM target) 8 KB
  __shared__ _Float16 Vt[64 * 64];        // V^T tile [d][key]             8 KB
  __shared__ _Float16 Pl[8][16 * 64];     // per-wave P^T [q][key]        16 KB

  const int lane = threadIdx.x & 31;
  const int wave = threadIdx.x >> 5;
  const int g    = lane >> 4;             // half-wave group
  const int ln   = lane & 15;
  const int h    = blockIdx.y;
  const int n    = blockIdx.z;
  const int qglob = blockIdx.x * 128 + wave * 16 + ln;   // this lane's query row

  const _Float16* khead = Kh + ((size_t)(n * HH + h)) * SS * DD;
  const _Float16* vhead = Vh + ((size_t)(n * HH + h)) * SS * DD;
  const _Float16* mrow  = Mh + (size_t)n * SS;
  // fold softmax scale AND log2(e) into Q: scores come out in log2 domain
  const float qscale = 0.044194173824159216f * 1.4426950408889634f;

  // Q^T B-fragments (lane column = this lane's query), f32 -> scaled f16 once
  const float* qrow = Qg + ((size_t)(n * SS + qglob)) * EE + h * DD;
  v16h qb0, qb1, ones;
#pragma unroll
  for (int i = 0; i < 16; ++i) {
    qb0[i]  = (_Float16)(qrow[(g << 3) + (i & 7) + ((i & 8) << 1)] * qscale);
    qb1[i]  = (_Float16)(qrow[32 + (g << 3) + (i & 7) + ((i & 8) << 1)] * qscale);
    ones[i] = (_Float16)1.0f;
  }

  v8f o[4];                                // O^T accumulator (64 d x 16 q)
#pragma unroll
  for (int ds = 0; ds < 4; ++ds) o[ds] = (v8f){};
  float rmax = -3.0e38f, rsum = 0.0f;

  // cooperative staging coords (V^T)
  const int skey = threadIdx.x >> 2;       // 0..63
  const int dch  = (threadIdx.x & 3) << 4; // 0,16,32,48
  const unsigned kl_lds = (unsigned)(uintptr_t)&Kl[0];

  for (int jb = 0; jb < SS; jb += 64) {
    // ---- stage K via TDM (wave 0) and V^T manually -------------------------
    __syncthreads();
    if (wave == 0)                         // TDM ignores EXEC; one issue/block
      tdm_load_tile(khead + (size_t)jb * DD, kl_lds);
    {
      const _Float16* vs = vhead + (size_t)(jb + skey) * DD + dch;
      v16h vv = *(const v16h*)vs;
#pragma unroll
      for (int i = 0; i < 16; ++i) Vt[(dch + i) * 64 + skey] = vv[i];
      if (jb + 64 < SS)                    // prefetch next V tile into L2
        __builtin_prefetch(vs + 64 * DD, 0, 1);
    }
    if (wave == 0)
      __builtin_amdgcn_s_wait_tensorcnt(0);
    __syncthreads();

    // ---- scores (log2 domain): S^T(key, q) = K . (qscale*Q)^T -------------
    v8f c[4];
#pragma unroll
    for (int grp = 0; grp < 4; ++grp) {
      const _Float16* kr = &Kl[(grp * 16 + ln) * 64];
      v16h ka_lo = cat8(*(const v8h*)(kr + (g << 3)),
                        *(const v8h*)(kr + 16 + (g << 3)));
      v16h ka_hi = cat8(*(const v8h*)(kr + 32 + (g << 3)),
                        *(const v8h*)(kr + 48 + (g << 3)));
      v8f cc = (v8f){};
      cc = WMMA_F16(ka_lo, qb0, cc);
      cc = WMMA_F16(ka_hi, qb1, cc);
      c[grp] = cc;
    }

    // ---- online softmax: all 32 values of a lane belong to ONE query ------
    float lmax = -3.0e38f;
#pragma unroll
    for (int grp = 0; grp < 4; ++grp)
#pragma unroll
      for (int r = 0; r < 8; ++r) lmax = fmaxf(lmax, c[grp][r]);
    lmax = fmaxf(lmax, __shfl_xor(lmax, 16, 32));   // join the two key-halves
    const float nm   = fmaxf(rmax, lmax);
    const float corr = exp2f(rmax - nm);            // raw v_exp_f32
    rmax = nm;

    // P^T -> LDS (C-layout rows are contiguous 8-half runs)
#pragma unroll
    for (int grp = 0; grp < 4; ++grp) {
      v8h ph;
#pragma unroll
      for (int r = 0; r < 8; ++r) ph[r] = (_Float16)exp2f(c[grp][r] - nm);
      *(v8h*)&Pl[wave][ln * 64 + grp * 16 + (g << 3)] = ph;
    }
    __builtin_amdgcn_wave_barrier();       // LDS is in-order within a wave

    // reload P^T as B-fragments and apply the mask multiplicatively
    v16h pf0 = *(const v16h*)&Pl[wave][ln * 64 + (g << 4)];
    v16h pf1 = *(const v16h*)&Pl[wave][ln * 64 + 32 + (g << 4)];
    pf0 *= *(const v16h*)(mrow + jb + (g << 4));
    pf1 *= *(const v16h*)(mrow + jb + 32 + (g << 4));

    // ---- O^T = V^T . P^T  +  row-sums via ones-WMMA -----------------------
    v8f st = (v8f){};
    st = WMMA_F16(ones, pf0, st);
    st = WMMA_F16(ones, pf1, st);
    rsum = rsum * corr + st[0];            // every slot holds sum(q = lane)

#pragma unroll
    for (int ds = 0; ds < 4; ++ds) {
      const _Float16* vr = &Vt[(ds * 16 + ln) * 64];
      v16h va_lo = cat8(*(const v8h*)(vr + (g << 4)),
                        *(const v8h*)(vr + 8 + (g << 4)));
      v16h va_hi = cat8(*(const v8h*)(vr + 32 + (g << 4)),
                        *(const v8h*)(vr + 40 + (g << 4)));
#pragma unroll
      for (int r = 0; r < 8; ++r) o[ds][r] *= corr;
      o[ds] = WMMA_F16(va_lo, pf0, o[ds]);
      o[ds] = WMMA_F16(va_hi, pf1, o[ds]);
    }
  }

  // ---- normalize; each lane owns one ctx row -> contiguous b128 stores ----
  const float inv = 1.0f / rsum;
  _Float16* crow = ctx + ((size_t)(n * SS + qglob)) * EE + h * DD;
#pragma unroll
  for (int ds = 0; ds < 4; ++ds) {
    v8h oh;
#pragma unroll
    for (int r = 0; r < 8; ++r) oh[r] = (_Float16)(o[ds][r] * inv);
    *(v8h*)(crow + ds * 16 + (g << 3)) = oh;
  }
}

// ---------------------------------------------------------------------------
// Kernel 2: out[m,o] = sum_k ctx[m,k] * W[o,k].
// 32 rows per block: two A-fragments share every W B-fragment (8 WMMA/K-step).
__global__ __launch_bounds__(256)
void out_proj_kernel(const _Float16* __restrict__ ctx, const _Float16* __restrict__ Wh,
                     float* __restrict__ out) {
  const int lane = threadIdx.x & 31;
  const int wave = threadIdx.x >> 5;
  const int g    = lane >> 4;
  const int ln   = lane & 15;
  const int mbase = blockIdx.x * 32;
  const int cb    = wave * 64;             // 8 waves cover 512 output cols

  v8f acc[2][4];
#pragma unroll
  for (int t = 0; t < 2; ++t)
#pragma unroll
    for (int j = 0; j < 4; ++j) acc[t][j] = (v8f){};

  for (int kt = 0; kt < EE; kt += 32) {
    const _Float16* ar0 = ctx + (size_t)(mbase + ln) * EE + kt;
    const _Float16* ar1 = ar0 + (size_t)16 * EE;
    v16h af0 = cat8(*(const v8h*)(ar0 + (g << 3)),
                    *(const v8h*)(ar0 + 16 + (g << 3)));
    v16h af1 = cat8(*(const v8h*)(ar1 + (g << 3)),
                    *(const v8h*)(ar1 + 16 + (g << 3)));
    const _Float16* wr = Wh + (size_t)(cb + ln) * EE + kt + (g << 3);
#pragma unroll
    for (int j = 0; j < 4; ++j) {
      v16h b = cat8(*(const v8h*)(wr + (size_t)(16 * j) * EE),
                    *(const v8h*)(wr + (size_t)(16 * j) * EE + 16));
      acc[0][j] = WMMA_F16(af0, b, acc[0][j]);
      acc[1][j] = WMMA_F16(af1, b, acc[1][j]);
    }
  }
#pragma unroll
  for (int t = 0; t < 2; ++t)
#pragma unroll
    for (int r = 0; r < 8; ++r) {
      float* orow = out + (size_t)(mbase + t * 16 + r + (g << 3)) * EE + cb + ln;
      orow[0]  = acc[t][0][r];
      orow[16] = acc[t][1][r];
      orow[32] = acc[t][2][r];
      orow[48] = acc[t][3][r];
    }
}

// ---------------------------------------------------------------------------
extern "C" void kernel_launch(void* const* d_in, const int* in_sizes, int n_in,
                              void* d_out, int out_size, void* d_ws, size_t ws_size,
                              hipStream_t stream) {
  const float* values = (const float*)d_in[0];
  const float* keys   = (const float*)d_in[1];
  const float* query  = (const float*)d_in[2];
  const int*   mask   = (const int*)d_in[3];
  const float* W_out  = (const float*)d_in[4];
  float* out = (float*)d_out;

  _Float16* ws  = (_Float16*)d_ws;          // ~49 MB of f16 scratch
  _Float16* Kh  = ws + KH_OFF;
  _Float16* Vh  = ws + VH_OFF;
  _Float16* ctx = ws + CTX_OFF;
  _Float16* Wh  = ws + WH_OFF;
  _Float16* Mh  = ws + MH_OFF;

  dim3 grid0(4096, 3);
  pack_kernel<<<grid0, 256, 0, stream>>>(values, keys, W_out, mask, Kh, Vh, Wh, Mh);

  dim3 grid1(SS / 128, HH, NN);             // (16, 8, 8)
  attn_fwd_kernel<<<grid1, 256, 0, stream>>>(Kh, Vh, query, Mh, ctx);

  dim3 grid2((NN * SS) / 32);               // 512 blocks
  out_proj_kernel<<<grid2, 256, 0, stream>>>(ctx, Wh, out);
}
